// GNN_node_Virtualnode_32684701122615
// MI455X (gfx1250) — matrix operand order repaired and checked by reading, not compiled, not run on hardware
//
#include <hip/hip_runtime.h>
#include <hip/hip_bf16.h>

#define LEAKY 0.1f

typedef __attribute__((ext_vector_type(16))) __bf16 bf16x16;
typedef __attribute__((ext_vector_type(8)))  __bf16 bf16x8;
typedef __attribute__((ext_vector_type(4)))  __bf16 bf16x4;
typedef __attribute__((ext_vector_type(8)))  float  f32x8;
typedef __attribute__((ext_vector_type(4)))  float  f32x4;

// ---------------------------------------------------------------------------
// Tiled bf16 WMMA GEMM:  C[M,N](ldc) = act( A[M,K](lda) @ B[K,N] + bias )
// B is supplied pre-transposed & pre-converted: Bt[n][k] bf16, pitch K.
// Block tile 64x128, K-step 32, 128 threads = 4 waves (wave32) in a 2x2 grid;
// each wave owns a 32x64 quadrant = 2x4 v_wmma_f32_16x16x32_bf16 tiles.
// Global->register->LDS pipeline: tile k+1 is fetched into VGPRs while tile k
// is consumed by WMMA.  Requires M%64==0, N%128==0, K%16==0 (all call sites
// satisfy this; the only K%32!=0 case is K=16 -> cold zero-padded path).
// ---------------------------------------------------------------------------
__global__ __launch_bounds__(128) void gemm_bf16(const float* __restrict__ A, int lda,
                                                 const __bf16* __restrict__ Bt,
                                                 const float* __restrict__ bias,
                                                 float* __restrict__ C, int ldc,
                                                 int K, int act)
{
  // 40-element row pitch (80 B = 5*16 B): keeps 16B alignment, spreads banks.
  __shared__ __attribute__((aligned(16))) __bf16 sA[64 * 40];    // sA[m][k]
  __shared__ __attribute__((aligned(16))) __bf16 sB[128 * 40];   // sB[n][k]

  const int tid  = threadIdx.x;
  const int m0   = blockIdx.y * 64;
  const int n0   = blockIdx.x * 128;
  const int wave = tid >> 5;
  const int lane = tid & 31;
  const int wm   = (wave >> 1) * 32;   // wave row within 64
  const int wn   = (wave & 1) * 64;    // wave col within 128
  const int lm   = lane & 15;
  const int hi   = lane >> 4;          // lane half: 0 or 1

  f32x8 acc[2][4] = {};
  f32x4 ar[4];
  bf16x8 brg[4];

  // --- staging helpers -----------------------------------------------------
  auto load_tile = [&](int k0) {
    if (K - k0 >= 32) {                       // fast path: full 32-wide K tile
      #pragma unroll
      for (int i = 0; i < 4; ++i) {           // A: 64x32 fp32 = 512 float4
        int e = tid + i * 128, r = e >> 3, kq = e & 7;
        ar[i] = *(const f32x4*)(A + (size_t)(m0 + r) * lda + k0 + kq * 4);
      }
      #pragma unroll
      for (int i = 0; i < 4; ++i) {           // Bt: 128 rows x 32 k bf16
        int e = tid + i * 128, n = e >> 2, kq = e & 3;
        brg[i] = *(const bf16x8*)(Bt + (size_t)(n0 + n) * K + k0 + kq * 8);
      }
    } else {                                  // cold path: zero-pad K remainder
      #pragma unroll
      for (int i = 0; i < 4; ++i) {
        int e = tid + i * 128, r = e >> 3, kq = e & 7;
        f32x4 v;
        #pragma unroll
        for (int j = 0; j < 4; ++j) {
          int kk = k0 + kq * 4 + j;
          v[j] = (kk < K) ? A[(size_t)(m0 + r) * lda + kk] : 0.f;
        }
        ar[i] = v;
      }
      #pragma unroll
      for (int i = 0; i < 4; ++i) {           // K%8==0 -> whole-chunk guard
        int e = tid + i * 128, n = e >> 2, kq = e & 3;
        bf16x8 v;
        if (k0 + kq * 8 < K) {
          v = *(const bf16x8*)(Bt + (size_t)(n0 + n) * K + k0 + kq * 8);
        } else {
          #pragma unroll
          for (int j = 0; j < 8; ++j) v[j] = (__bf16)0.f;
        }
        brg[i] = v;
      }
    }
  };

  auto store_tile = [&]() {
    #pragma unroll
    for (int i = 0; i < 4; ++i) {             // A -> sA[m][k], packed b64 stores
      int e = tid + i * 128, r = e >> 3, kq = e & 7;
      bf16x4 p;
      p[0] = (__bf16)ar[i][0]; p[1] = (__bf16)ar[i][1];
      p[2] = (__bf16)ar[i][2]; p[3] = (__bf16)ar[i][3];
      *(bf16x4*)&sA[r * 40 + kq * 4] = p;
    }
    #pragma unroll
    for (int i = 0; i < 4; ++i) {             // Bt -> sB[n][k], b128 stores
      int e = tid + i * 128, n = e >> 2, kq = e & 3;
      *(bf16x8*)&sB[n * 40 + kq * 8] = brg[i];
    }
  };

  // --- main pipeline -------------------------------------------------------
  load_tile(0);
  for (int k0 = 0; k0 < K; k0 += 32) {
    __syncthreads();                 // previous tile's LDS reads complete
    store_tile();
    __syncthreads();                 // tile visible to all waves
    if (k0 + 32 < K) load_tile(k0 + 32);   // prefetch next tile into VGPRs

    // A fragments: ISA 16-bit A 16x32 layout. lanes 0-15 -> K {0..7,16..23},
    // lanes 16-31 -> K {8..15,24..31}.
    bf16x16 af[2], bfr[4];
    const int kha = hi ? 8 : 0;
    #pragma unroll
    for (int mt = 0; mt < 2; ++mt) {
      int row = wm + mt * 16 + lm;
      bf16x8 lo = *(const bf16x8*)&sA[row * 40 + kha];
      bf16x8 hv = *(const bf16x8*)&sA[row * 40 + kha + 16];
      af[mt] = __builtin_shufflevector(lo, hv, 0,1,2,3,4,5,6,7,8,9,10,11,12,13,14,15);
    }
    // B fragments: lanes 0-15 -> K {0..7,16..23}, lanes 16-31 -> K {8..15,24..31}.
    const int khb = hi ? 8 : 0;
    #pragma unroll
    for (int nt = 0; nt < 4; ++nt) {
      int col = wn + nt * 16 + lm;
      bf16x8 lo = *(const bf16x8*)&sB[col * 40 + khb];
      bf16x8 hv = *(const bf16x8*)&sB[col * 40 + khb + 16];
      bfr[nt] = __builtin_shufflevector(lo, hv, 0,1,2,3,4,5,6,7,8,9,10,11,12,13,14,15);
    }

    #pragma unroll
    for (int mt = 0; mt < 2; ++mt)
      #pragma unroll
      for (int nt = 0; nt < 4; ++nt)
        acc[mt][nt] = __builtin_amdgcn_wmma_f32_16x16x32_bf16(
            /*neg_a=*/false, af[mt], /*neg_b=*/false, bfr[nt],
            /*c_mod=*/(short)0, acc[mt][nt], /*reuse_a=*/false, /*reuse_b=*/false);
  }

  // Epilogue: C/D layout -> row = vg + 8*hi, col = lm within each 16x16 tile.
  #pragma unroll
  for (int mt = 0; mt < 2; ++mt) {
    #pragma unroll
    for (int nt = 0; nt < 4; ++nt) {
      int col = n0 + wn + nt * 16 + lm;
      float bv = bias ? bias[col] : 0.f;
      #pragma unroll
      for (int vg = 0; vg < 8; ++vg) {
        int row = m0 + wm + mt * 16 + vg + 8 * hi;
        float y = acc[mt][nt][vg] + bv;
        if (act) y = (y < 0.f) ? y * LEAKY : y;
        C[(size_t)row * ldc + col] = y;
      }
    }
  }
}

// ---------------------------------------------------------------------------
// Transpose + fp32->bf16 convert:  out[N][K] bf16 = in[K][N] fp32 transposed.
// Tiled through LDS; both global sides coalesced.  block (32,8), grid
// ((N+31)/32, (K+31)/32).
// ---------------------------------------------------------------------------
__global__ __launch_bounds__(256) void transpose_bf16_k(const float* __restrict__ in,
                                                        __bf16* __restrict__ out,
                                                        int K, int N) {
  __shared__ __bf16 tile[32][33];
  int kt = blockIdx.y * 32, nt = blockIdx.x * 32;
  int tx = threadIdx.x, ty = threadIdx.y;
  #pragma unroll
  for (int j = 0; j < 32; j += 8) {
    int k = kt + ty + j, n = nt + tx;
    tile[ty + j][tx] = (k < K && n < N) ? (__bf16)in[(size_t)k * N + n] : (__bf16)0.f;
  }
  __syncthreads();
  #pragma unroll
  for (int j = 0; j < 32; j += 8) {
    int n = nt + ty + j, k = kt + tx;
    if (n < N && k < K) out[(size_t)n * K + k] = tile[tx][ty + j];
  }
}

// ---------------------------------------------------------------------------
// Elementwise / reduction kernels
// ---------------------------------------------------------------------------
__global__ void init_vn_k(const float* __restrict__ emb, float* __restrict__ vn) {
  vn[blockIdx.x * 256 + threadIdx.x] = emb[threadIdx.x];
}

// h[i,:] += vn[batch[i],:]   (in place, matching reference) and mirror into
// hcat column block 0 (concat input for mlp1 + pooling source).
__global__ void add_vn_k(const float* __restrict__ vn, const int* __restrict__ batch,
                         float* __restrict__ h, float* __restrict__ hcat) {
  int i = blockIdx.x, c = threadIdx.x;
  float v = h[(size_t)i * 256 + c] + vn[(size_t)batch[i] * 256 + c];
  h[(size_t)i * 256 + c] = v;
  hcat[(size_t)i * 768 + c] = v;
}

__global__ void zero_k(float* __restrict__ p, int n) {
  int i = blockIdx.x * 256 + threadIdx.x;
  if (i < n) p[i] = 0.f;
}

__global__ void pool_acc_k(const float* __restrict__ hcat, const int* __restrict__ batch,
                           float* __restrict__ pooled, float* __restrict__ cnt) {
  int i = blockIdx.x, c = threadIdx.x;
  int v = batch[i];
  atomicAdd(&pooled[(size_t)v * 256 + c], hcat[(size_t)i * 768 + c]);
  if (c == 0) atomicAdd(&cnt[v], 1.f);
}

__global__ void vn_in_k(const float* __restrict__ pooled, const float* __restrict__ cnt,
                        const float* __restrict__ vn, float* __restrict__ vnin) {
  int v = blockIdx.x, c = threadIdx.x;
  float d = fmaxf(cnt[v], 1.f);
  vnin[v * 256 + c] = pooled[v * 256 + c] / d + vn[v * 256 + c];
}

__global__ void vn_accum_k(const float* __restrict__ g2, float* __restrict__ vn) {
  int i = blockIdx.x * 256 + threadIdx.x;
  vn[i] += g2[i];
}

// dst = leaky( LayerNorm(src) * g + b ) + res   (one 256-wide row per block)
__global__ __launch_bounds__(256) void ln_act_res_k(const float* __restrict__ src,
                                                    const float* __restrict__ g,
                                                    const float* __restrict__ b,
                                                    const float* __restrict__ res,
                                                    float* __restrict__ dst) {
  __shared__ float red[256];
  int r = blockIdx.x, t = threadIdx.x;
  float x = src[(size_t)r * 256 + t];
  red[t] = x; __syncthreads();
  for (int o = 128; o > 0; o >>= 1) { if (t < o) red[t] += red[t + o]; __syncthreads(); }
  float m = red[0] * (1.f / 256.f);
  __syncthreads();
  float d = x - m;
  red[t] = d * d; __syncthreads();
  for (int o = 128; o > 0; o >>= 1) { if (t < o) red[t] += red[t + o]; __syncthreads(); }
  float var = red[0] * (1.f / 256.f);
  float y = g[t] * d * rsqrtf(var + 1e-5f) + b[t];
  y = (y < 0.f) ? y * LEAKY : y;
  dst[(size_t)r * 256 + t] = y + res[(size_t)r * 256 + t];
}

// ---------------------------------------------------------------------------
// Host-side orchestration
// ---------------------------------------------------------------------------
// Dims (fixed by the reference): N_INST=8192, N_NET=4096, E=256, ND=32,
// NETD=16, L=3, NVN=64.
//
// Input index map. Top-level in setup_inputs() insertion order:
//   0:x 1:x_net 2:net_inst_adj 3:drive 4:sink 5:batch 6:num_vn
// params flattened as a JAX pytree (dict keys sorted alphabetically):
//   7:enc1.w 8:enc1.b 9:enc2.w 10:enc2.b 11:enc_net1.w 12:enc_net1.b
//   13:enc_net2.w 14:enc_net2.b
//   layers l=0..2 at base 15+l*14:
//     +0 ln_b +1 ln_g +2 mlp1.w +3 mlp1.b +4 mlp2.w +5 mlp2.b
//     +6 phi1.w +7 phi1.b +8 phi2.w +9 phi2.b +10 psi1.w +11 psi1.b
//     +12 psi2.w +13 psi2.b
//   57: vn_emb
//   vn_mlp l=0..1 at base 58+l*4: +0 m1.w +1 m1.b +2 m2.w +3 m2.b
extern "C" void kernel_launch(void* const* d_in, const int* in_sizes, int n_in,
                              void* d_out, int out_size, void* d_ws, size_t ws_size,
                              hipStream_t stream) {
  const float* x        = (const float*)d_in[0];
  const float* x_net    = (const float*)d_in[1];
  const float* net_inst = (const float*)d_in[2];
  const float* drive    = (const float*)d_in[3];
  const float* sink     = (const float*)d_in[4];
  const int*   batch    = (const int*)d_in[5];
  auto P = [&](int i) { return (const float*)d_in[i]; };

  // -------- fp32 workspace (floats).  region0 overlays t0 (encoder temp),
  // ta/tb (psi temps) and m1out (mlp1 output) -- lifetimes are disjoint.
  float* W = (float*)d_ws;
  size_t off = 0;
  float* region0 = W;              off += 6291456;    // max(t0 4.19M, m1out 6.29M)
  float* t0      = region0;
  float* ta      = region0;
  float* tb      = region0 + 2097152;
  float* m1out   = region0;
  float* h       = W + off;        off += 2097152;    // [8192,256]
  float* hn      = W + off;        off += 1048576;    // [4096,256]
  float* ntmp    = W + off;        off += 1048576;    // [4096,256]
  float* net_agg = W + off;        off += 1048576;    // [4096,256]
  float* hcat    = W + off;        off += 6291456;    // [8192,768]
  float* hc      = W + off;        off += 2097152;    // [8192,256]
  float* net_out = W + off;        off += 1048576;    // [4096,256]
  float* vn      = W + off;        off += 16384;      // [64,256]
  float* pooled  = W + off;        off += 16384;      // [64,256]
  float* cnt     = W + off;        off += 64;
  float* vnin    = W + off;        off += 16384;
  float* g1      = W + off;        off += 32768;      // [64,512]
  float* g2      = W + off;        off += 16384;      // [64,256]

  // -------- bf16 workspace: transposed B operands (Bt[n][k], pitch K).
  __bf16* B16 = (__bf16*)(W + off);
  size_t boff = 0;
  auto balloc = [&](size_t n) { __bf16* p = B16 + boff; boff += n; return p; };
  __bf16* enc1t = balloc(16384);    // [512][32]
  __bf16* enc2t = balloc(131072);   // [256][512]
  __bf16* ent1t = balloc(4096);     // [256][16]
  __bf16* ent2t = balloc(65536);    // [256][256]
  __bf16 *lp1t[3], *lp2t[3], *ls1t[3], *ls2t[3], *lm1t[3], *lm2t[3];
  for (int l = 0; l < 3; ++l) {
    lp1t[l] = balloc(65536);  lp2t[l] = balloc(65536);
    ls1t[l] = balloc(65536);  ls2t[l] = balloc(65536);
    lm1t[l] = balloc(589824);                 // [768][768]
    lm2t[l] = balloc(196608);                 // [256][768]
  }
  __bf16 *vm1t[2], *vm2t[2];
  for (int l = 0; l < 2; ++l) { vm1t[l] = balloc(131072); vm2t[l] = balloc(131072); }
  __bf16* nat = balloc(1048576);    // net_agg^T  [256][4096]
  __bf16* hct = balloc(2097152);    // hc^T       [256][8192]

  if (ws_size < off * sizeof(float) + boff * sizeof(__bf16)) return;  // not enough scratch

  auto gemm = [&](const float* A, int lda, const __bf16* Bt, const float* bias,
                  float* C, int ldc, int M, int N, int K, int act) {
    gemm_bf16<<<dim3(N / 128, M / 64), dim3(128), 0, stream>>>(A, lda, Bt, bias, C, ldc, K, act);
  };
  auto xpose = [&](const float* in, __bf16* outT, int K, int N) {
    transpose_bf16_k<<<dim3((N + 31) / 32, (K + 31) / 32), dim3(32, 8), 0, stream>>>(in, outT, K, N);
  };

  // One-time (per launch) weight transposes to bf16.
  xpose(P(7), enc1t, 32, 512);
  xpose(P(9), enc2t, 512, 256);
  xpose(P(11), ent1t, 16, 256);
  xpose(P(13), ent2t, 256, 256);
  for (int l = 0; l < 3; ++l) {
    const int LB = 15 + l * 14;
    xpose(P(LB + 6),  lp1t[l], 256, 256);
    xpose(P(LB + 8),  lp2t[l], 256, 256);
    xpose(P(LB + 10), ls1t[l], 256, 256);
    xpose(P(LB + 12), ls2t[l], 256, 256);
    xpose(P(LB + 2),  lm1t[l], 768, 768);
    xpose(P(LB + 4),  lm2t[l], 768, 256);
  }
  for (int l = 0; l < 2; ++l) {
    const int VB = 58 + l * 4;
    xpose(P(VB + 0), vm1t[l], 256, 512);
    xpose(P(VB + 2), vm2t[l], 512, 256);
  }

  // Encoders.
  gemm(x,     32,  enc1t, P(8),  t0,  512, 8192, 512,  32, 1);   // lr(x@enc1+b)
  gemm(t0,    512, enc2t, P(10), h,   256, 8192, 256, 512, 1);   // lr(..@enc2+b)
  gemm(x_net, 16,  ent1t, P(12), ntmp,256, 4096, 256,  16, 1);   // lr(x_net@enc_net1+b)
  gemm(ntmp,  256, ent2t, P(14), hn,  256, 4096, 256, 256, 1);   // lr(..@enc_net2+b)
  init_vn_k<<<64, 256, 0, stream>>>(P(57), vn);

  for (int l = 0; l < 3; ++l) {
    const int LB = 15 + l * 14;
    const float *ln_b = P(LB + 0), *ln_g = P(LB + 1);
    const float *m1b = P(LB + 3), *m2b = P(LB + 5);
    const float *p1b = P(LB + 7), *p2b = P(LB + 9);
    const float *s1b = P(LB + 11), *s2b = P(LB + 13);

    // h += vn[batch] (in place), mirrored into hcat[:,0:256].
    add_vn_k<<<8192, 256, 0, stream>>>(vn, batch, h, hcat);

    // net_agg = phi(hn)
    gemm(hn,   256, lp1t[l], p1b, ntmp,    256, 4096, 256, 256, 1);
    gemm(ntmp, 256, lp2t[l], p2b, net_agg, 256, 4096, 256, 256, 0);
    xpose(net_agg, nat, 4096, 256);

    // h_drive = drive @ net_agg  -> hcat[:,256:512]
    gemm(drive, 4096, nat, nullptr, hcat + 256, 768, 8192, 256, 4096, 0);
    // h_sink = psi(sink @ net_agg) -> hcat[:,512:768]
    gemm(sink, 4096, nat, nullptr, ta, 256, 8192, 256, 4096, 0);
    gemm(ta, 256, ls1t[l], s1b, tb, 256, 8192, 256, 256, 1);
    gemm(tb, 256, ls2t[l], s2b, hcat + 512, 768, 8192, 256, 256, 0);

    // hc = mlp(concat)
    gemm(hcat,  768, lm1t[l], m1b, m1out, 768, 8192, 768, 768, 1);
    gemm(m1out, 768, lm2t[l], m2b, hc,    256, 8192, 256, 768, 0);

    // net_out = net_inst_adj @ hc
    xpose(hc, hct, 8192, 256);
    gemm(net_inst, 8192, hct, nullptr, net_out, 256, 4096, 256, 8192, 0);

    // LayerNorm + LeakyReLU + residual (shared LN params). Final hn -> d_out.
    float* hn_dst = (l == 2) ? (float*)d_out : hn;
    ln_act_res_k<<<4096, 256, 0, stream>>>(net_out, ln_g, ln_b, hn, hn_dst);
    ln_act_res_k<<<8192, 256, 0, stream>>>(hc, ln_g, ln_b, h, h);

    if (l < 2) {
      zero_k<<<(16384 + 64 + 255) / 256, 256, 0, stream>>>(pooled, 16384 + 64);
      pool_acc_k<<<8192, 256, 0, stream>>>(hcat, batch, pooled, cnt);  // uses h_list[layer]=h+vn
      vn_in_k<<<64, 256, 0, stream>>>(pooled, cnt, vn, vnin);
      gemm(vnin, 256, vm1t[l], P(58 + l * 4 + 1), g1, 512, 64, 512, 256, 1);
      gemm(g1,   512, vm2t[l], P(58 + l * 4 + 3), g2, 256, 64, 256, 512, 1);
      vn_accum_k<<<64, 256, 0, stream>>>(g2, vn);
    }
  }
}
